// RgnModel_44040594653184
// MI455X (gfx1250) — matrix operand order repaired and checked by reading, not compiled, not run on hardware
//
#include <hip/hip_runtime.h>
#include <hip/hip_bf16.h>

// ---------------- problem constants ----------------
constexpr int T_LEN = 512;
constexpr int BATCH = 32;
constexpr int EMB   = 41;
constexpr int HID   = 800;
constexpr int G4    = 4 * HID;   // 3200 gate columns
constexpr int UNITS = 20;
constexpr int NPAD  = 32;        // padded head output width (UNITS -> 32)
constexpr int OUTD  = 2 * HID;   // 1600 bilstm output width
constexpr int KX0   = 64;        // EMB+1=42 padded to 64 (mult of 32)
constexpr int KX1   = OUTD;      // 1600

// persistent LSTM kernel geometry: 2dir * 2mtile * 200ntile = 800 wave-tiles
constexpr int NBLK    = 100;
constexpr int THREADS = 256;     // 8 waves/block -> exactly 800 waves

typedef __bf16 bf16_t;
typedef __attribute__((ext_vector_type(16))) __bf16 v16bf;
typedef __attribute__((ext_vector_type(8)))  float  v8f;

union Frag { unsigned u[8]; v16bf v; };

// ---- K-chunked bf16 WMMA accumulation over a 16x16 tile --------------------
// Ap: this lane's A row base (row = M index)
// Bp: this lane's B row base (row = N index)
// Fragment index math per ISA 7.12.2 (16-bit A 16x32 / B 32x16 layouts).
template <int K>
__device__ __forceinline__ v8f bf16_gemm_acc(const bf16_t* __restrict__ Ap,
                                             const bf16_t* __restrict__ Bp,
                                             int sel, v8f acc)
{
  #pragma unroll 5
  for (int kk = 0; kk < K; kk += 32) {
    Frag fa, fb;
    #pragma unroll
    for (int v = 0; v < 8; ++v) {
      const int ka = kk + ((v & 4) ? 16 : 0) + sel * 8 + 2 * (v & 3);
      fa.u[v] = *(const unsigned*)(Ap + ka);
      const int kb = kk + ((v & 4) ? 16 : 0) + sel * 8 + 2 * (v & 3);
      fb.u[v] = *(const unsigned*)(Bp + kb);
    }
    acc = __builtin_amdgcn_wmma_f32_16x16x32_bf16(false, fa.v, false, fb.v,
                                                  (short)0, acc, false, false);
  }
  return acc;
}

// ---------------- grid-wide barrier (agent scope, monotonic counter) --------
__device__ __forceinline__ void grid_bar(unsigned* cnt, unsigned& epoch) {
  __threadfence();
  __syncthreads();
  if (threadIdx.x == 0) {
    __hip_atomic_fetch_add(cnt, 1u, __ATOMIC_ACQ_REL, __HIP_MEMORY_SCOPE_AGENT);
    const unsigned target = (unsigned)NBLK * epoch;
    while (__hip_atomic_load(cnt, __ATOMIC_ACQUIRE, __HIP_MEMORY_SCOPE_AGENT) < target)
      __builtin_amdgcn_s_sleep(1);
  }
  ++epoch;
  __syncthreads();
}

// ---------------- prep kernels ----------------
__global__ void cvt_bf16_kernel(const float* __restrict__ s, bf16_t* __restrict__ d, long n) {
  long i = (long)blockIdx.x * blockDim.x + threadIdx.x;
  if (i < n) d[i] = (bf16_t)s[i];
}

// w_ih0 (2,3200,42) f32 -> (2,3200,64) bf16, zero padded
__global__ void pad_wih0_kernel(const float* __restrict__ s, bf16_t* __restrict__ d) {
  long i = (long)blockIdx.x * blockDim.x + threadIdx.x;
  const long total = 2L * G4 * KX0;
  if (i >= total) return;
  int  k = (int)(i % KX0);
  long r = i / KX0;
  d[i] = (k < EMB + 1) ? (bf16_t)s[r * (EMB + 1) + k] : (bf16_t)0.0f;
}

// lin_w (20,1600) f32 -> (32,1600) bf16, rows 20..31 zero
__global__ void pad_linw_kernel(const float* __restrict__ s, bf16_t* __restrict__ d) {
  long i = (long)blockIdx.x * blockDim.x + threadIdx.x;
  const long total = (long)NPAD * OUTD;
  if (i >= total) return;
  int u = (int)(i / OUTD);
  d[i] = (u < UNITS) ? (bf16_t)s[i] : (bf16_t)0.0f;
}

// sequences (T,B,41) -> xcat (T,B,64) bf16 with col 41 = t, 42..63 = 0
__global__ void xcat_kernel(const float* __restrict__ seq, bf16_t* __restrict__ d) {
  long i = (long)blockIdx.x * blockDim.x + threadIdx.x;
  const long total = (long)T_LEN * BATCH * KX0;
  if (i >= total) return;
  int  k = (int)(i % KX0);
  long r = i / KX0;                 // t*B + b
  int  t = (int)(r / BATCH);
  float v = (k < EMB) ? seq[r * EMB + k] : ((k == EMB) ? (float)t : 0.0f);
  d[i] = (bf16_t)v;
}

// ---------------- persistent bidirectional LSTM layer ----------------
// gates(32 x 3200) = [h_{t-1} | x_t] * [Wh | Wx]^T + b  (per direction)
// v_wmma_f32_16x16x32_bf16, f32 accumulation (bias folded into C init).
template <int KX>
__global__ void __launch_bounds__(THREADS)
lstm_layer_kernel(const bf16_t* __restrict__ Wh,    // (2, G4, HID)   bf16
                  const bf16_t* __restrict__ Wx,    // (2, G4, KX)    bf16
                  const bf16_t* __restrict__ Xbf,   // (T, B, KX)     bf16
                  const float*  __restrict__ bias,  // (2, G4)        f32
                  bf16_t* __restrict__ Out,         // (T, B, 1600)   bf16
                  bf16_t* __restrict__ Hprev,       // (2, B, HID)    bf16 (zeroed)
                  float*  __restrict__ Cst,         // (2, B, HID)    f32  (zeroed)
                  float*  __restrict__ Gbuf,        // (2, B, G4)     f32 scratch
                  unsigned* __restrict__ barcnt)    // zeroed
{
  const int lane = threadIdx.x & 31;
  const int wave = blockIdx.x * (THREADS / 32) + (threadIdx.x >> 5); // 0..799
  const int dir  = wave / 400;
  const int rem  = wave % 400;
  const int mt   = rem / 200;          // 0..1   (16-row batch tile)
  const int nt   = rem % 200;          // 0..199 (16-col gate tile)
  const int sel  = lane >> 4;          // K-half select (ISA 7.12.2)
  const int l15  = lane & 15;

  const int arow = mt * 16 + l15;      // A row (batch index)
  const int ncol = nt * 16 + l15;      // B row / D col (gate column)

  const bf16_t* WhRow = Wh + (size_t)dir * G4 * HID + (size_t)ncol * HID;
  const bf16_t* WxRow = Wx + (size_t)dir * G4 * KX  + (size_t)ncol * KX;
  const bf16_t* Hrow  = Hprev + (size_t)dir * BATCH * HID + (size_t)arow * HID;
  float*        Gd    = Gbuf  + (size_t)dir * BATCH * G4;
  const float   bcol  = bias[(size_t)dir * G4 + ncol];   // per-column bias

  unsigned epoch = 1;

  for (int t = 0; t < T_LEN; ++t) {
    const int teff = dir ? (T_LEN - 1 - t) : t;

    // ---------- WMMA GEMM phase (bias pre-loaded into C) ----------
    v8f acc;
    #pragma unroll
    for (int r = 0; r < 8; ++r) acc[r] = bcol;

    acc = bf16_gemm_acc<HID>(Hrow, WhRow, sel, acc);                   // 25 wmma
    const bf16_t* Xt = Xbf + ((size_t)teff * BATCH + arow) * KX;
    acc = bf16_gemm_acc<KX>(Xt, WxRow, sel, acc);                      // 2 / 50 wmma

    // prefetch next step's x row for this lane (global_prefetch_b8)
    if (t + 1 < T_LEN) {
      const int tn = dir ? (T_LEN - 2 - t) : (t + 1);
      __builtin_prefetch(Xbf + ((size_t)tn * BATCH + arow) * KX + sel * (KX / 2), 0, 3);
    }

    // D tile: VGPR r -> row M = r + 8*sel, col N = l15 (ISA 7.12.2)
    #pragma unroll
    for (int r = 0; r < 8; ++r) {
      const int m = mt * 16 + r + 8 * sel;
      Gd[(size_t)m * G4 + ncol] = acc[r];
    }

    grid_bar(barcnt, epoch);   // gates fully written

    // ---------- elementwise LSTM cell ----------
    for (int e = blockIdx.x * THREADS + threadIdx.x; e < 2 * BATCH * HID;
         e += NBLK * THREADS) {
      const int d2 = e / (BATCH * HID);
      const int r2 = e % (BATCH * HID);
      const int b2 = r2 / HID;
      const int u  = r2 % HID;
      const float* G2 = Gbuf + (size_t)d2 * BATCH * G4 + (size_t)b2 * G4;
      const float gi = G2[u];
      const float gf = G2[HID + u];
      const float gg = G2[2 * HID + u];
      const float go = G2[3 * HID + u];
      const float si = 1.0f / (1.0f + __expf(-gi));
      const float sf = 1.0f / (1.0f + __expf(-gf));
      const float so = 1.0f / (1.0f + __expf(-go));
      float* cp = Cst + (size_t)d2 * BATCH * HID + (size_t)b2 * HID + u;
      const float c = sf * (*cp) + si * tanhf(gg);
      *cp = c;
      const float h = so * tanhf(c);
      const int te2 = d2 ? (T_LEN - 1 - t) : t;
      Hprev[(size_t)d2 * BATCH * HID + (size_t)b2 * HID + u] = (bf16_t)h;
      Out[((size_t)te2 * BATCH + b2) * OUTD + d2 * HID + u]  = (bf16_t)h;
    }

    grid_bar(barcnt, epoch);   // h_t visible before next step's GEMM
  }
}

// ---------------- head GEMM: logits(16384 x 32) = H1 * lin_w_pad^T ---------
// M = T*B (1024 tiles), N = 32 (2 tiles), K = 1600 (50 wmma per tile).
__global__ void __launch_bounds__(256)
head_gemm_kernel(const bf16_t* __restrict__ H1,      // (T*B, 1600) bf16
                 const bf16_t* __restrict__ Wl,      // (32, 1600)  bf16 padded
                 const float*  __restrict__ lin_b,   // (20,)
                 float* __restrict__ Lg)             // (T*B, 32)   f32
{
  const int lane = threadIdx.x & 31;
  const int wave = blockIdx.x * 8 + (threadIdx.x >> 5);   // 0..2047
  const int mt   = wave >> 1;          // 0..1023
  const int nt   = wave & 1;           // 0..1
  const int sel  = lane >> 4;
  const int l15  = lane & 15;

  const int arow = mt * 16 + l15;      // row in T*B
  const int ncol = nt * 16 + l15;      // logit unit (padded)

  v8f acc;
  const float bc = (ncol < UNITS) ? lin_b[ncol] : 0.0f;
  #pragma unroll
  for (int r = 0; r < 8; ++r) acc[r] = bc;

  acc = bf16_gemm_acc<OUTD>(H1 + (size_t)arow * OUTD,
                            Wl + (size_t)ncol * OUTD, sel, acc);

  #pragma unroll
  for (int r = 0; r < 8; ++r) {
    const int m = mt * 16 + r + 8 * sel;
    Lg[(size_t)m * NPAD + nt * 16 + l15] = acc[r];
  }
}

// ---------------- softmax + angle mix: one wave per (t,b) ------------------
__global__ void __launch_bounds__(256)
angle_kernel(const float* __restrict__ Lg, const float* __restrict__ alph,
             float* __restrict__ Ang)
{
  const int wv   = blockIdx.x * 8 + (threadIdx.x >> 5);   // t*B + b
  const int lane = threadIdx.x & 31;
  if (wv >= T_LEN * BATCH) return;

  const float logit = (lane < UNITS) ? Lg[(size_t)wv * NPAD + lane] : -INFINITY;
  float m = logit;
  #pragma unroll
  for (int o = 16; o > 0; o >>= 1) m = fmaxf(m, __shfl_xor(m, o, 32));
  float e = (lane < UNITS) ? __expf(logit - m) : 0.0f;
  float sum = e;
  #pragma unroll
  for (int o = 16; o > 0; o >>= 1) sum += __shfl_xor(sum, o, 32);
  const float p = e / sum;

  #pragma unroll
  for (int j = 0; j < 3; ++j) {
    const float a = (lane < UNITS) ? alph[lane * 3 + j] : 0.0f;
    float sn = p * __sinf(a);
    float cs = p * __cosf(a);
    #pragma unroll
    for (int o = 16; o > 0; o >>= 1) {
      sn += __shfl_xor(sn, o, 32);
      cs += __shfl_xor(cs, o, 32);
    }
    if (lane == 0) Ang[(size_t)wv * 3 + j] = atan2f(sn, cs);
  }
}

// ---------------- sequential geometric unit: one thread per batch ----------
__global__ void geom_kernel(const float* __restrict__ Ang, float* __restrict__ Out)
{
  const int b = threadIdx.x;
  if (b >= BATCH) return;
  const float BL[3] = {1.329f, 1.459f, 1.525f};
  const float BA[3] = {2.034f, 2.119f, 1.937f};
  float Ax = 0.f, Ay = 0.f, Az = 1.f;
  float Bx = 0.f, By = 1.f, Bz = 0.f;
  float Cx = 1.f, Cy = 0.f, Cz = 0.f;
  Out[(0 * BATCH + b) * 3 + 0] = Ax; Out[(0 * BATCH + b) * 3 + 1] = Ay; Out[(0 * BATCH + b) * 3 + 2] = Az;
  Out[(1 * BATCH + b) * 3 + 0] = Bx; Out[(1 * BATCH + b) * 3 + 1] = By; Out[(1 * BATCH + b) * 3 + 2] = Bz;
  Out[(2 * BATCH + b) * 3 + 0] = Cx; Out[(2 * BATCH + b) * 3 + 1] = Cy; Out[(2 * BATCH + b) * 3 + 2] = Cz;

  for (int t = 1; t < T_LEN; ++t) {
    #pragma unroll
    for (int i = 0; i < 3; ++i) {
      const float P  = Ang[((size_t)t * BATCH + b) * 3 + i];
      const float R  = BL[i], Ta = BA[i];
      const float d0 = -R * __cosf(Ta);
      const float d1 =  R * __cosf(P) * __sinf(Ta);
      const float d2 =  R * __sinf(P) * __sinf(Ta);
      float bcx = Cx - Bx, bcy = Cy - By, bcz = Cz - Bz;
      float il  = rsqrtf(bcx * bcx + bcy * bcy + bcz * bcz);
      bcx *= il; bcy *= il; bcz *= il;
      const float abx = Bx - Ax, aby = By - Ay, abz = Bz - Az;
      float nx = aby * bcz - abz * bcy;
      float ny = abz * bcx - abx * bcz;
      float nz = abx * bcy - aby * bcx;
      il = rsqrtf(nx * nx + ny * ny + nz * nz);
      nx *= il; ny *= il; nz *= il;
      const float mx = ny * bcz - nz * bcy;  // cross(n, bc)
      const float my = nz * bcx - nx * bcz;
      const float mz = nx * bcy - ny * bcx;
      const float Dx = bcx * d0 + mx * d1 + nx * d2 + Cx;
      const float Dy = bcy * d0 + my * d1 + ny * d2 + Cy;
      const float Dz = bcz * d0 + mz * d1 + nz * d2 + Cz;
      Ax = Bx; Ay = By; Az = Bz;
      Bx = Cx; By = Cy; Bz = Cz;
      Cx = Dx; Cy = Dy; Cz = Dz;
      const int row = 3 + 3 * (t - 1) + i;
      Out[((size_t)row * BATCH + b) * 3 + 0] = Dx;
      Out[((size_t)row * BATCH + b) * 3 + 1] = Dy;
      Out[((size_t)row * BATCH + b) * 3 + 2] = Dz;
    }
  }
}

// ---------------- host launcher ----------------
static inline size_t align_up(size_t x) { return (x + 255) & ~size_t(255); }

extern "C" void kernel_launch(void* const* d_in, const int* in_sizes, int n_in,
                              void* d_out, int out_size, void* d_ws, size_t ws_size,
                              hipStream_t stream) {
  (void)in_sizes; (void)n_in; (void)out_size; (void)ws_size;
  const float* seq    = (const float*)d_in[0];
  const float* w_ih0  = (const float*)d_in[2];   // (2,3200,42)
  const float* w_hh0  = (const float*)d_in[3];   // (2,3200,800)
  const float* b0     = (const float*)d_in[4];   // (2,3200)
  const float* w_ih1  = (const float*)d_in[5];   // (2,3200,1600)
  const float* w_hh1  = (const float*)d_in[6];   // (2,3200,800)
  const float* b1     = (const float*)d_in[7];   // (2,3200)
  const float* lin_w  = (const float*)d_in[8];   // (20,1600)
  const float* lin_b  = (const float*)d_in[9];   // (20,)
  const float* alph   = (const float*)d_in[10];  // (20,3)

  // workspace layout (~155 MB)
  size_t off = 0;
  const size_t off_bar  = off; off += align_up(256);
  const size_t off_c    = off; off += align_up((size_t)2 * BATCH * HID * 4);
  const size_t off_h    = off; off += align_up((size_t)2 * BATCH * HID * 2);
  const size_t state_bytes = off;                      // memset region per layer
  const size_t off_g    = off; off += align_up((size_t)2 * BATCH * G4 * 4);
  const size_t off_x    = off; off += align_up((size_t)T_LEN * BATCH * KX0 * 2);
  const size_t off_wih0 = off; off += align_up((size_t)2 * G4 * KX0 * 2);
  const size_t off_whh0 = off; off += align_up((size_t)2 * G4 * HID * 2);
  const size_t off_wih1 = off; off += align_up((size_t)2 * G4 * KX1 * 2);
  const size_t off_whh1 = off; off += align_up((size_t)2 * G4 * HID * 2);
  const size_t off_o0   = off; off += align_up((size_t)T_LEN * BATCH * OUTD * 2);
  const size_t off_o1   = off; off += align_up((size_t)T_LEN * BATCH * OUTD * 2);
  const size_t off_wl   = off; off += align_up((size_t)NPAD * OUTD * 2);
  const size_t off_lg   = off; off += align_up((size_t)T_LEN * BATCH * NPAD * 4);
  const size_t off_ang  = off; off += align_up((size_t)T_LEN * BATCH * 3 * 4);

  char* W = (char*)d_ws;
  unsigned* barcnt = (unsigned*)(W + off_bar);
  float*    cst    = (float*)   (W + off_c);
  bf16_t*   hprev  = (bf16_t*)  (W + off_h);
  float*    gbuf   = (float*)   (W + off_g);
  bf16_t*   xcat   = (bf16_t*)  (W + off_x);
  bf16_t*   wih0bf = (bf16_t*)  (W + off_wih0);
  bf16_t*   whh0bf = (bf16_t*)  (W + off_whh0);
  bf16_t*   wih1bf = (bf16_t*)  (W + off_wih1);
  bf16_t*   whh1bf = (bf16_t*)  (W + off_whh1);
  bf16_t*   out0bf = (bf16_t*)  (W + off_o0);
  bf16_t*   out1bf = (bf16_t*)  (W + off_o1);
  bf16_t*   wlbf   = (bf16_t*)  (W + off_wl);
  float*    lgbuf  = (float*)   (W + off_lg);
  float*    ang    = (float*)   (W + off_ang);

  auto blocks = [](long n) { return (unsigned)((n + 255) / 256); };

  // prep: zero state, build bf16 operands
  hipMemsetAsync(W, 0, state_bytes, stream);
  xcat_kernel<<<blocks((long)T_LEN * BATCH * KX0), 256, 0, stream>>>(seq, xcat);
  pad_wih0_kernel<<<blocks(2L * G4 * KX0), 256, 0, stream>>>(w_ih0, wih0bf);
  cvt_bf16_kernel<<<blocks(2L * G4 * HID), 256, 0, stream>>>(w_hh0, whh0bf, 2L * G4 * HID);
  cvt_bf16_kernel<<<blocks(2L * G4 * KX1), 256, 0, stream>>>(w_ih1, wih1bf, 2L * G4 * KX1);
  cvt_bf16_kernel<<<blocks(2L * G4 * HID), 256, 0, stream>>>(w_hh1, whh1bf, 2L * G4 * HID);
  pad_linw_kernel<<<blocks((long)NPAD * OUTD), 256, 0, stream>>>(lin_w, wlbf);

  // layer 0 (K = 800 + 64)
  lstm_layer_kernel<KX0><<<NBLK, THREADS, 0, stream>>>(whh0bf, wih0bf, xcat, b0,
                                                       out0bf, hprev, cst, gbuf, barcnt);
  // layer 1 (K = 800 + 1600)
  hipMemsetAsync(W, 0, state_bytes, stream);
  lstm_layer_kernel<KX1><<<NBLK, THREADS, 0, stream>>>(whh1bf, wih1bf, out0bf, b1,
                                                       out1bf, hprev, cst, gbuf, barcnt);

  // head GEMM on WMMA: 2048 wave-tiles
  head_gemm_kernel<<<256, 256, 0, stream>>>(out1bf, wlbf, lin_b, lgbuf);

  // softmax + angle mixing
  angle_kernel<<<(T_LEN * BATCH) / 8, 256, 0, stream>>>(lgbuf, alph, ang);

  // sequential geometric extension
  geom_kernel<<<1, 32, 0, stream>>>(ang, (float*)d_out);
}
